// RibonanzaNetPairwiseMixer_17325898072633
// MI455X (gfx1250) — matrix operand order repaired and checked by reading, not compiled, not run on hardware
//
#include <hip/hip_runtime.h>
#include <math.h>

// ---------------------------------------------------------------------------
// RibonanzaNet pairwise mixer (triangle multiplicative update, outgoing)
// B=2, L=512, C=64.  Mixed precision: bf16 WMMA (16x16x32, f32 accum).
// Triangle GEMM stages operand tiles with the Tensor Data Mover (TDM) when the
// toolchain exposes __builtin_amdgcn_tensor_load_to_lds, double-buffered and
// synchronized with s_wait_tensorcnt; otherwise falls back to manual staging.
//
// Workspace layout (bytes):
//   [0      .. 40960)   wt5    : 5 x 64 x 64 bf16, [p][c_out][c_in]
//   [40960  .. 49152)   wtout  : 64 x 64 bf16,     [c_out][c_in]
//   [64K    .. +64M)    left   : bf16 [b][c][i][k]
//   [+64M   .. +128M)   right  : bf16 [b][c][j][k]
//   [+128M  .. +192M)   ogate  : bf16 [b][c][i][j]
//   [+192M  .. +320M)   tri    : f32  [b][c][i][j]
// ---------------------------------------------------------------------------

typedef __attribute__((ext_vector_type(16))) __bf16 v16bf;
typedef __attribute__((ext_vector_type(8)))  float  v8f;

union FragU { unsigned int u[8]; v16bf v; };

// A-matrix fragment (16x32 bf16, M x K), tile row-major [16][kpitch halves].
__device__ __forceinline__ v16bf load_frag_row16(const __bf16* tile, int kpitchHalves) {
  int lane = threadIdx.x & 31;
  int m = lane & 15;
  int b0 = (lane < 16) ? 0 : 8; // dword offset of K base (0 or 16 halves)
  const unsigned int* p = (const unsigned int*)(tile + m * kpitchHalves);
  FragU f;
  f.u[0] = p[b0 + 0]; f.u[1] = p[b0 + 1]; f.u[2] = p[b0 + 2]; f.u[3] = p[b0 + 3];
  f.u[4] = p[b0 + 4]; f.u[5] = p[b0 + 5]; f.u[6] = p[b0 + 6]; f.u[7] = p[b0 + 7];
  return f.v;
}

// B-matrix fragment (32x16 bf16, K x N), tile row-major [N=16][kpitch halves].
__device__ __forceinline__ v16bf load_frag_col16(const __bf16* tile, int kpitchHalves) {
  int lane = threadIdx.x & 31;
  int n = lane & 15;
  int b0 = (lane < 16) ? 0 : 8; // dword offset of K base (0 or 16 halves)
  const unsigned int* p = (const unsigned int*)(tile + n * kpitchHalves);
  FragU f;
#pragma unroll
  for (int v = 0; v < 8; ++v) f.u[v] = p[b0 + v];
  return f.v;
}

__device__ __forceinline__ v8f wmma_bf16(v16bf a, v16bf b, v8f c) {
  return __builtin_amdgcn_wmma_f32_16x16x32_bf16(false, a, false, b, (short)0, c,
                                                 false, false);
}

__device__ __forceinline__ float sigmoidf_(float x) {
  return 1.0f / (1.0f + __expf(-x));
}

// ---------------------------------------------------------------------------
// Tensor Data Mover helper: load a [rows x 32] bf16 tile (row stride 512
// elements) from global into LDS with 1-dword padding every 16 dwords, giving
// a 34-half (68 B) LDS row pitch == the bank-conflict-free fragment pitch.
// ---------------------------------------------------------------------------
#if defined(__gfx1250__) && __has_builtin(__builtin_amdgcn_tensor_load_to_lds)
#define HAVE_TDM 1
typedef __attribute__((ext_vector_type(4))) unsigned int u32x4;
typedef __attribute__((ext_vector_type(4))) int          i32x4;
typedef __attribute__((ext_vector_type(8))) int          i32x8;

__device__ __forceinline__ void tdm_load_tile(const __bf16* gsrc,
                                              unsigned int ldsAddr, int rows) {
  unsigned long long ga = (unsigned long long)(uintptr_t)gsrc;
  u32x4 g0;
  g0[0] = 1u;                                             // count=1, user mode
  g0[1] = ldsAddr;                                        // lds_addr (bytes)
  g0[2] = (unsigned int)(ga & 0xffffffffu);               // global_addr lo
  g0[3] = (unsigned int)((ga >> 32) & 0x1ffffffu)         // global_addr hi
          | (2u << 30);                                   // type=2 ("image")
  i32x8 g1;
  g1[0] = (int)((1u << 16)      // data_size = 1 -> 2 bytes
              | (1u << 20)      // pad_enable
              | (3u << 22));    // pad_interval = 3 -> every 16 dwords (64 B)
                                // pad_amount  = 0 -> 1 dword (4 B)
  g1[1] = (int)(512u << 16);    // tensor_dim0 = 512 (k extent), low 16 bits
  g1[2] = (int)(512u << 16);    // tensor_dim0 hi=0 | tensor_dim1 = 512 (rows)
  g1[3] = (int)(32u << 16);     // tensor_dim1 hi=0 | tile_dim0 = 32 (k chunk)
  g1[4] = rows;                 // tile_dim1 = rows, tile_dim2 = 0 (2D)
  g1[5] = 512;                  // tensor_dim0_stride lo (elements)
  g1[6] = 0;                    // stride hi | tensor_dim1_stride lo (unused, 2D)
  g1[7] = 0;
  i32x4 z4 = {0, 0, 0, 0};
#if __has_include(<hip/amd_detail/amd_gfx1250_TDM.h>)
  i32x8 z8 = {0, 0, 0, 0, 0, 0, 0, 0};
  __builtin_amdgcn_tensor_load_to_lds(g0, g1, z4, z4, z8, 0);
#else
  __builtin_amdgcn_tensor_load_to_lds(g0, g1, z4, z4, 0);
#endif
}
#else
#define HAVE_TDM 0
#endif

// ---------------------------------------------------------------------------
// Kernel 0: weights f32 -> bf16, transposed to [c_out][c_in]
// ---------------------------------------------------------------------------
__global__ __launch_bounds__(256) void prep_weights(
    const float* Wlp, const float* Wrp, const float* Wlg, const float* Wrg,
    const float* Wog, const float* Wout, __bf16* wt5, __bf16* wtout) {
  const float* Ws[5] = {Wlp, Wrp, Wlg, Wrg, Wog};
  int t = threadIdx.x;
  for (int p = 0; p < 5; ++p)
    for (int idx = t; idx < 4096; idx += 256) {
      int k = idx >> 6, n = idx & 63;          // W[k][n] -> wt[p][n][k]
      wt5[(p * 64 + n) * 64 + k] = (__bf16)Ws[p][idx];
    }
  for (int idx = t; idx < 4096; idx += 256) {
    int k = idx >> 6, n = idx & 63;
    wtout[n * 64 + k] = (__bf16)Wout[idx];
  }
}

// ---------------------------------------------------------------------------
// Kernel 1: fused input LayerNorm + 5 projections + mask + sigmoid gating.
// ---------------------------------------------------------------------------
__global__ __launch_bounds__(256) void ln_proj_kernel(
    const float* __restrict__ x, const float* __restrict__ mask,
    const float* __restrict__ lnw, const float* __restrict__ lnb,
    const float* __restrict__ blp, const float* __restrict__ brp,
    const float* __restrict__ blg, const float* __restrict__ brg,
    const float* __restrict__ bog, const __bf16* __restrict__ wt5g,
    __bf16* __restrict__ left_s, __bf16* __restrict__ right_s,
    __bf16* __restrict__ og_s) {
  __shared__ __align__(16) __bf16 WtS[5][64][66];  // 42240 B
  __shared__ __align__(16) __bf16 Axh[8][16][66];  // 16896 B

  int tid = threadIdx.x, lane = tid & 31, wv = tid >> 5;

  for (int idx = tid; idx < 5 * 4096; idx += 256) {
    int p = idx >> 12, r = idx & 4095, n = r >> 6, k = r & 63;
    WtS[p][n][k] = wt5g[idx];
  }

  long long tileT = (long long)blockIdx.x * 8 + wv;
  long long row0  = tileT * 16;            // 16 rows share (b, p), q spans 16
  int b  = (int)(row0 >> 18);
  int pI = (int)((row0 >> 9) & 511);
  int q0 = (int)(row0 & 511);

  for (int r = 0; r < 16; ++r) {
    const float2 v = ((const float2*)(x + (size_t)(row0 + r) * 64))[lane];
    float s = v.x + v.y;
    for (int o = 16; o > 0; o >>= 1) s += __shfl_xor(s, o, 32);
    float mu = s * (1.0f / 64.0f);
    float dx = v.x - mu, dy = v.y - mu;
    float ss = dx * dx + dy * dy;
    for (int o = 16; o > 0; o >>= 1) ss += __shfl_xor(ss, o, 32);
    float rs = rsqrtf(ss * (1.0f / 64.0f) + 1e-5f);
    Axh[wv][r][2 * lane]     = (__bf16)(dx * rs * lnw[2 * lane]     + lnb[2 * lane]);
    Axh[wv][r][2 * lane + 1] = (__bf16)(dy * rs * lnw[2 * lane + 1] + lnb[2 * lane + 1]);
  }
  __syncthreads();

  v16bf xb0 = load_frag_col16(&Axh[wv][0][0], 66);   // K = 0..31  (c_in)
  v16bf xb1 = load_frag_col16(&Axh[wv][0][32], 66);  // K = 32..63
  float maskv = mask[b * 512 + pI];
  int hi = lane >> 4, nlo = lane & 15;

  for (int t = 0; t < 4; ++t) {                      // c_out tiles
    v8f alp = {}, arp = {}, alg = {}, arg_ = {}, aog = {};
    v16bf wa0, wa1;
#define PROJ(P, ACC)                                            \
    wa0 = load_frag_row16(&WtS[P][t * 16][0], 66);              \
    wa1 = load_frag_row16(&WtS[P][t * 16][32], 66);             \
    ACC = wmma_bf16(wa0, xb0, ACC);                             \
    ACC = wmma_bf16(wa1, xb1, ACC);
    PROJ(0, alp) PROJ(1, arp) PROJ(2, alg) PROJ(3, arg_) PROJ(4, aog)
#undef PROJ
#pragma unroll
    for (int v = 0; v < 8; ++v) {
      int c = t * 16 + v + hi * 8;                   // M coordinate = c_out
      size_t gi = (((size_t)(b * 64 + c)) << 18) + ((size_t)pI << 9) +
                  (size_t)(q0 + nlo);                // [b][c][row][k]
      float l = (alp[v] + blp[c]) * maskv * sigmoidf_(alg[v]  + blg[c]);
      float r = (arp[v] + brp[c]) * maskv * sigmoidf_(arg_[v] + brg[c]);
      float g = sigmoidf_(aog[v] + bog[c]);
      left_s[gi]  = (__bf16)l;
      right_s[gi] = (__bf16)r;
      og_s[gi]    = (__bf16)g;
    }
  }
}

// ---------------------------------------------------------------------------
// Kernel 2: triangle einsum. Per (b,c): tri = L @ R^T (512^3 GEMM).
// Block tile 128(i) x 64(j); 8 waves, each computing a 32x32 output block
// (4 accumulators, 4 frag loads -> 4 WMMAs per 32-K chunk).  K chunks are
// staged by the TDM into double-buffered LDS (overlapped with compute).
// ---------------------------------------------------------------------------
__global__ __launch_bounds__(256) void triangle_kernel(
    const __bf16* __restrict__ left_s, const __bf16* __restrict__ right_s,
    float* __restrict__ tri) {
  __shared__ __align__(16) __bf16 Al[2][128][34];  // 17408 B, 68 B row pitch
  __shared__ __align__(16) __bf16 Bl[2][64][34];   //  8704 B

  int tid = threadIdx.x, lane = tid & 31, wv = tid >> 5;
  int bc = blockIdx.z;                       // b*64 + c
  int i0 = blockIdx.y * 128, j0 = blockIdx.x * 64;
  size_t base = ((size_t)bc) << 18;          // * L*L
  int ti = (wv >> 1) * 32;                   // 0..96  (i offset in block tile)
  int tj = (wv & 1) * 32;                    // 0 or 32 (j offset)
  int hi = lane >> 4, nlo = lane & 15;

  const __bf16* gL = left_s  + base + ((size_t)i0 << 9);
  const __bf16* gR = right_s + base + ((size_t)j0 << 9);

  v8f a00 = {}, a01 = {}, a10 = {}, a11 = {};

#if HAVE_TDM
  unsigned int ldsA[2] = {(unsigned int)(uintptr_t)&Al[0][0][0],
                          (unsigned int)(uintptr_t)&Al[1][0][0]};
  unsigned int ldsB[2] = {(unsigned int)(uintptr_t)&Bl[0][0][0],
                          (unsigned int)(uintptr_t)&Bl[1][0][0]};
  // prologue: DMA chunk 0 into buffer 0
  if (wv == 0)      tdm_load_tile(gL, ldsA[0], 128);
  else if (wv == 1) tdm_load_tile(gR, ldsB[0], 64);
  if (wv < 2) __builtin_amdgcn_s_wait_tensorcnt(0);
  __syncthreads();

  for (int kc = 0; kc < 16; ++kc) {
    int cur = kc & 1;
    if (kc < 15) {                           // async prefetch of next chunk
      int k1 = (kc + 1) * 32;
      if (wv == 0)      tdm_load_tile(gL + k1, ldsA[cur ^ 1], 128);
      else if (wv == 1) tdm_load_tile(gR + k1, ldsB[cur ^ 1], 64);
    }
    v16bf A0 = load_frag_row16(&Al[cur][ti][0], 34);
    v16bf A1 = load_frag_row16(&Al[cur][ti + 16][0], 34);
    v16bf B0 = load_frag_col16(&Bl[cur][tj][0], 34);
    v16bf B1 = load_frag_col16(&Bl[cur][tj + 16][0], 34);
    a00 = wmma_bf16(A0, B0, a00);
    a01 = wmma_bf16(A0, B1, a01);
    a10 = wmma_bf16(A1, B0, a10);
    a11 = wmma_bf16(A1, B1, a11);
    if (wv < 2) __builtin_amdgcn_s_wait_tensorcnt(0);
    __syncthreads();
  }
#else
  // fallback: synchronous VGPR staging
  for (int k0 = 0; k0 < 512; k0 += 32) {
    __syncthreads();
    for (int idx = tid; idx < 192 * 16; idx += 256) {
      int r = idx >> 4, d = idx & 15;
      if (r < 128)
        ((unsigned int*)&Al[0][r][0])[d] =
            ((const unsigned int*)(gL + ((size_t)r << 9) + k0))[d];
      else
        ((unsigned int*)&Bl[0][r - 128][0])[d] =
            ((const unsigned int*)(gR + ((size_t)(r - 128) << 9) + k0))[d];
    }
    __syncthreads();
    v16bf A0 = load_frag_row16(&Al[0][ti][0], 34);
    v16bf A1 = load_frag_row16(&Al[0][ti + 16][0], 34);
    v16bf B0 = load_frag_col16(&Bl[0][tj][0], 34);
    v16bf B1 = load_frag_col16(&Bl[0][tj + 16][0], 34);
    a00 = wmma_bf16(A0, B0, a00);
    a01 = wmma_bf16(A0, B1, a01);
    a10 = wmma_bf16(A1, B0, a10);
    a11 = wmma_bf16(A1, B1, a11);
  }
#endif

#pragma unroll
  for (int v = 0; v < 8; ++v) {
    int iA = i0 + ti + v + hi * 8;
    int iB = iA + 16;
    int jA = j0 + tj + nlo, jB = jA + 16;
    size_t rA = base + ((size_t)iA << 9);
    size_t rB = base + ((size_t)iB << 9);
    tri[rA + jA] = a00[v];                   // lanes contiguous in j
    tri[rA + jB] = a01[v];
    tri[rB + jA] = a10[v];
    tri[rB + jB] = a11[v];
  }
}

// ---------------------------------------------------------------------------
// Kernel 3: output LayerNorm * gate, W_out projection, +bias +residual.
// ---------------------------------------------------------------------------
__global__ __launch_bounds__(128) void out_kernel(
    const float* __restrict__ tri, const __bf16* __restrict__ og_s,
    const float* __restrict__ lnw, const float* __restrict__ lnb,
    const __bf16* __restrict__ wtoutg, const float* __restrict__ bout,
    const float* __restrict__ x_res, float* __restrict__ out) {
  __shared__ __align__(16) float  Tt[128 * 65];   // 33280 B
  __shared__ __align__(16) __bf16 Gx[128 * 66];   // 16896 B
  __shared__ __align__(16) __bf16 WoS[64][66];    //  8448 B

  int tid = threadIdx.x, lane = tid & 31, wv = tid >> 5;
  int b = blockIdx.z;
  int i0 = blockIdx.y * 8, j0 = blockIdx.x * 16;
  int ii = tid >> 4, jj = tid & 15;
  size_t posBase = ((size_t)(i0 + ii) << 9) + (size_t)(j0 + jj);

  for (int idx = tid; idx < 4096; idx += 128)
    WoS[idx >> 6][idx & 63] = wtoutg[idx];

  for (int c = 0; c < 64; ++c) {
    size_t g = (((size_t)(b * 64 + c)) << 18) + posBase;
    Tt[tid * 65 + c] = tri[g];
    Gx[tid * 66 + c] = og_s[g];
  }

  float s = 0.0f;
  for (int c = 0; c < 64; ++c) s += Tt[tid * 65 + c];
  float mu = s * (1.0f / 64.0f);
  float ss = 0.0f;
  for (int c = 0; c < 64; ++c) {
    float d = Tt[tid * 65 + c] - mu;
    ss += d * d;
  }
  float rs = rsqrtf(ss * (1.0f / 64.0f) + 1e-5f);
  for (int c = 0; c < 64; ++c) {
    float xh = (Tt[tid * 65 + c] - mu) * rs * lnw[c] + lnb[c];
    float g  = (float)Gx[tid * 66 + c];
    Gx[tid * 66 + c] = (__bf16)(xh * g);
  }
  __syncthreads();

  int hi = lane >> 4, nlo = lane & 15;
  for (int rt = wv * 2; rt < wv * 2 + 2; ++rt) {
    v16bf a0 = load_frag_row16(&Gx[(rt * 16) * 66], 66);
    v16bf a1 = load_frag_row16(&Gx[(rt * 16) * 66 + 32], 66);
    for (int nt = 0; nt < 4; ++nt) {
      v16bf w0 = load_frag_col16(&WoS[nt * 16][0], 66);
      v16bf w1 = load_frag_col16(&WoS[nt * 16][32], 66);
      v8f acc = {};
      acc = wmma_bf16(a0, w0, acc);
      acc = wmma_bf16(a1, w1, acc);
#pragma unroll
      for (int v = 0; v < 8; ++v) {
        int pos = rt * 16 + v + hi * 8;
        int c = nt * 16 + nlo;
        size_t rowg = ((((size_t)b << 9) + i0 + (pos >> 4)) << 9) + j0 + (pos & 15);
        size_t o = rowg * 64 + c;
        out[o] = acc[v] + bout[c] + x_res[o];
      }
    }
  }
}

// ---------------------------------------------------------------------------
extern "C" void kernel_launch(void* const* d_in, const int* in_sizes, int n_in,
                              void* d_out, int out_size, void* d_ws, size_t ws_size,
                              hipStream_t stream) {
  const float* x    = (const float*)d_in[0];
  const float* mask = (const float*)d_in[1];
  const float* lnw  = (const float*)d_in[2];
  const float* lnb  = (const float*)d_in[3];
  const float* Wlp  = (const float*)d_in[4];  const float* blp = (const float*)d_in[5];
  const float* Wrp  = (const float*)d_in[6];  const float* brp = (const float*)d_in[7];
  const float* Wlg  = (const float*)d_in[8];  const float* blg = (const float*)d_in[9];
  const float* Wrg  = (const float*)d_in[10]; const float* brg = (const float*)d_in[11];
  const float* Wog  = (const float*)d_in[12]; const float* bog = (const float*)d_in[13];
  const float* lnow = (const float*)d_in[14]; const float* lnob = (const float*)d_in[15];
  const float* Wout = (const float*)d_in[16]; const float* bout = (const float*)d_in[17];
  (void)in_sizes; (void)n_in; (void)out_size; (void)ws_size;

  char* ws = (char*)d_ws;
  const size_t MB64 = 67108864ull;
  __bf16* wt5     = (__bf16*)(ws);
  __bf16* wtout   = (__bf16*)(ws + 40960);
  __bf16* left_s  = (__bf16*)(ws + 65536);
  __bf16* right_s = (__bf16*)(ws + 65536 + MB64);
  __bf16* og_s    = (__bf16*)(ws + 65536 + 2 * MB64);
  float*  tri     = (float*) (ws + 65536 + 3 * MB64);

  prep_weights<<<1, 256, 0, stream>>>(Wlp, Wrp, Wlg, Wrg, Wog, Wout, wt5, wtout);
  ln_proj_kernel<<<4096, 256, 0, stream>>>(x, mask, lnw, lnb, blp, brp, blg, brg,
                                           bog, wt5, left_s, right_s, og_s);
  triangle_kernel<<<dim3(8, 4, 128), 256, 0, stream>>>(left_s, right_s, tri);
  out_kernel<<<dim3(32, 64, 2), 128, 0, stream>>>(tri, og_s, lnow, lnob, wtout,
                                                  bout, x, (float*)d_out);
}
